// KHWl_HGNN_28166395527443
// MI455X (gfx1250) — matrix operand order repaired and checked by reading, not compiled, not run on hardware
//
#include <hip/hip_runtime.h>

typedef __attribute__((ext_vector_type(16))) _Float16 v16h;
typedef __attribute__((ext_vector_type(8)))  _Float16 v8h;
typedef __attribute__((ext_vector_type(4)))  _Float16 v4h;
typedef __attribute__((ext_vector_type(8)))  float    v8f;

// ---------------- problem sizes ----------------
#define NN     50000
#define MM     25000
#define NNZ0   400000
#define SUBN   60000
#define SUBM   30000
#define SUBNNZ 240000
#define SS     20000
#define KT     3
#define KHM    10000
#define KHNNZ  80000
#define NG     64
#define FT     128
#define HID    256
#define NCLS   10
#define SXW    (FT + 1 + HID)   // 385 logical concat width
#define SXP    416              // padded to 13*32 for aligned, tail-free GEMM

__device__ __forceinline__ void atomAddF(float* p, float v) {
  __hip_atomic_fetch_add(p, v, __ATOMIC_RELAXED, __HIP_MEMORY_SCOPE_AGENT);
}
__device__ __forceinline__ int clampi(int v, int hi) { return v < hi ? v : hi; }

// ---------------- elementwise / sparse kernels ----------------
__global__ void k_fill(float* __restrict__ p, float v, long n) {
  long i = (long)blockIdx.x * blockDim.x + threadIdx.x;
  long st = (long)gridDim.x * blockDim.x;
  for (; i < n; i += st) p[i] = v;
}

__global__ void k_copy(float* __restrict__ dst, const float* __restrict__ src, long n) {
  long i = (long)blockIdx.x * blockDim.x + threadIdx.x;
  long st = (long)gridDim.x * blockDim.x;
  for (; i < n; i += st) dst[i] = src[i];
}

__global__ void k_degree(const int* __restrict__ rows, const float* __restrict__ vals,
                         float* __restrict__ deg, int nnz) {
  int i = blockIdx.x * blockDim.x + threadIdx.x;
  if (i < nnz) atomAddF(&deg[rows[i]], vals[i]);
}

__global__ void k_normvals(const int* __restrict__ rows, const float* __restrict__ vals,
                           const float* __restrict__ deg, float* __restrict__ out, int nnz) {
  int i = blockIdx.x * blockDim.x + threadIdx.x;
  if (i < nnz) {
    float d = deg[rows[i]];
    out[i] = vals[i] * (d == 0.0f ? 1.0f : 1.0f / d);
  }
}

// out[rows[e], :] += vals[e] * X[cols[e], :]   (one workgroup per nnz)
__global__ void k_spmm(const int* __restrict__ rows, const int* __restrict__ cols,
                       const float* __restrict__ vals, const float* __restrict__ X,
                       int ldx, float* __restrict__ out, int ldo, int F, int nnz) {
  int e = blockIdx.x;
  if (e >= nnz) return;
  int r = rows[e], c = cols[e];
  float v = vals[e];
  const float* xr = X + (size_t)c * ldx;
  float* orow = out + (size_t)r * ldo;
  if ((F & 3) == 0 && ((ldx | ldo) & 3) == 0) {
    for (int f = threadIdx.x * 4; f < F; f += blockDim.x * 4) {
      float4 xv = *(const float4*)(xr + f);
      atomAddF(&orow[f + 0], v * xv.x);
      atomAddF(&orow[f + 1], v * xv.y);
      atomAddF(&orow[f + 2], v * xv.z);
      atomAddF(&orow[f + 3], v * xv.w);
    }
  } else {
    for (int f = threadIdx.x; f < F; f += blockDim.x)
      atomAddF(&orow[f], v * xr[f]);
  }
}

__global__ void k_setcol(float* __restrict__ dst, int ld, int col,
                         const float* __restrict__ src, int n) {
  int i = blockIdx.x * blockDim.x + threadIdx.x;
  if (i < n) dst[(size_t)i * ld + col] = src[i];
}

// out[seg[i], :] += X[i, :]
__global__ void k_segsum_rows(const int* __restrict__ seg, const float* __restrict__ X,
                              int F, float* __restrict__ out, int rowsN) {
  int i = blockIdx.x;
  if (i >= rowsN) return;
  int s = seg[i];
  const float* xr = X + (size_t)i * F;
  float* orow = out + (size_t)s * F;
  for (int f = threadIdx.x; f < F; f += blockDim.x)
    atomAddF(&orow[f], xr[f]);
}

__global__ void k_relu_add(float* __restrict__ dst, const float* __restrict__ a,
                           const float* __restrict__ b, long n) {
  long i = (long)blockIdx.x * blockDim.x + threadIdx.x;
  long st = (long)gridDim.x * blockDim.x;
  for (; i < n; i += st) {
    float v = a[i] + (b ? b[i] : 0.0f);
    dst[i] = fmaxf(v, 0.0f);
  }
}

__global__ void k_copycols(float* __restrict__ dst, int ldd, int off,
                           const float* __restrict__ src, int lds, int F, int rowsN) {
  int i = blockIdx.x;
  if (i >= rowsN) return;
  for (int f = threadIdx.x; f < F; f += blockDim.x)
    dst[(size_t)i * ldd + off + f] = src[(size_t)i * lds + f];
}

// dst[s, off+f] = mean_k x[kset[s*KT+k], f]
__global__ void k_gather_mean(const int* __restrict__ kset, const float* __restrict__ x,
                              int ldx, float* __restrict__ dst, int ldd, int off,
                              int F, int rowsN) {
  int s = blockIdx.x;
  if (s >= rowsN) return;
  int i0 = kset[s * KT + 0], i1 = kset[s * KT + 1], i2 = kset[s * KT + 2];
  for (int f = threadIdx.x; f < F; f += blockDim.x) {
    float acc = x[(size_t)i0 * ldx + f] + x[(size_t)i1 * ldx + f] + x[(size_t)i2 * ldx + f];
    dst[(size_t)s * ldd + off + f] = acc * (1.0f / (float)KT);
  }
}

__global__ void k_count(const int* __restrict__ b, float* __restrict__ cnt, int n) {
  int i = blockIdx.x * blockDim.x + threadIdx.x;
  if (i < n) atomAddF(&cnt[b[i]], 1.0f);
}

__global__ void k_div(float* __restrict__ out, const float* __restrict__ sums,
                      const float* __restrict__ cnt, int n, int F) {
  int i = blockIdx.x * blockDim.x + threadIdx.x;
  if (i < n * F) out[i] = sums[i] / fmaxf(cnt[i / F], 1.0f);
}

// ---------------- WMMA GEMM: C = relu?(A @ B + bias) ----------------
// Block tile 32x128, BK=32; 8 waves, each wave owns a 16x32 strip
// (one A fragment reused across two 16x16x32 WMMAs).
// LDS: A row-major (fragment = 2 contiguous 8-half runs),
//      B transposed (fragment = 2 contiguous 8-half runs per column).
// Tail paths use clamped addresses + mask-MULTIPLY (not select) so clang
// cannot sink the loads into exec-guarded regions.
#define BM 32
#define BN 128
#define BK 32
#define LDA 40    // halves per As row (32 + 8 pad)
#define LDBT 40   // halves per BsT row (32 + 8 pad)

__launch_bounds__(256, 2)
__global__ void k_gemm_wmma(const float* __restrict__ A, int lda,
                            const float* __restrict__ B,
                            const float* __restrict__ bias, float* __restrict__ C,
                            int Mrows, int Ncols, int Kdim, int relu) {
  __shared__ _Float16 As[BM * LDA];     //  2560 B
  __shared__ _Float16 BsT[BN * LDBT];   // 10240 B

  const int tid  = threadIdx.x;
  const int lane = tid & 31;
  const int wave = tid >> 5;
  const int wm   = wave >> 2;    // 0..1
  const int wn   = wave & 3;     // 0..3 (32-col strip)
  const int g    = lane >> 4;
  const int r    = lane & 15;

  const int rowBase = blockIdx.y * BM;
  const int colBase = blockIdx.x * BN;
  const bool kvec   = ((lda & 3) == 0);

  // per-thread staging coordinates (loop-invariant)
  const int ar = tid >> 3;              // A: row in tile 0..31
  const int ac = (tid & 7) * 4;         // A: col in tile 0,4,..28
  const int a_gr = rowBase + ar;
  const int a_cr = clampi(a_gr, Mrows - 1);        // clamped row
  const bool a_rok = (a_gr < Mrows);

  const int b_col = tid & (BN - 1);     // B: column 0..127
  const int b_kb  = (tid >> 7) * 16;    // B: k sub-block 0 or 16
  const int b_gc  = colBase + b_col;
  const int b_cc  = clampi(b_gc, Ncols - 1);       // clamped col
  const bool b_cok = (b_gc < Ncols);

  v8f acc0 = {}, acc1 = {};

  for (int k0 = 0; k0 < Kdim; k0 += BK) {
    // ---- stage A tile (BM x BK) ----
    {
      int gc = k0 + ac;
      v4h h;
      if (kvec && a_rok && gc + 3 < Kdim) {
        float4 f = *(const float4*)(A + (size_t)a_gr * lda + gc);
        h[0] = (_Float16)f.x; h[1] = (_Float16)f.y;
        h[2] = (_Float16)f.z; h[3] = (_Float16)f.w;
      } else {
        const float* arow_p = A + (size_t)a_cr * lda;
        #pragma unroll
        for (int u = 0; u < 4; ++u) {
          int c  = gc + u;
          int cc = clampi(c, Kdim - 1);
          float f = arow_p[cc];                       // unconditional load
          float m = (a_rok && c < Kdim) ? 1.0f : 0.0f;
          h[u] = (_Float16)(f * m);                   // mask-mul: no load sinking
        }
      }
      *(v4h*)&As[ar * LDA + ac] = h;                  // ds_store_b64
    }
    // ---- stage B tile (BK x BN) transposed: 16 K-values of one column ----
    {
      v8h lo, hi;
      if (b_cok && k0 + b_kb + 15 < Kdim) {
        const float* bp = B + (size_t)(k0 + b_kb) * Ncols + b_gc;
        #pragma unroll
        for (int j = 0; j < 8; ++j) lo[j] = (_Float16)bp[(size_t)j * Ncols];
        bp += (size_t)8 * Ncols;
        #pragma unroll
        for (int j = 0; j < 8; ++j) hi[j] = (_Float16)bp[(size_t)j * Ncols];
      } else {
        #pragma unroll
        for (int j = 0; j < 8; ++j) {
          int kk  = k0 + b_kb + j;
          int kkc = clampi(kk, Kdim - 1);
          float f = B[(size_t)kkc * Ncols + b_cc];    // unconditional load
          float m = (b_cok && kk < Kdim) ? 1.0f : 0.0f;
          lo[j] = (_Float16)(f * m);
        }
        #pragma unroll
        for (int j = 0; j < 8; ++j) {
          int kk  = k0 + b_kb + 8 + j;
          int kkc = clampi(kk, Kdim - 1);
          float f = B[(size_t)kkc * Ncols + b_cc];
          float m = (b_cok && kk < Kdim) ? 1.0f : 0.0f;
          hi[j] = (_Float16)(f * m);
        }
      }
      *(v8h*)&BsT[b_col * LDBT + b_kb]     = lo;      // ds_store_b128
      *(v8h*)&BsT[b_col * LDBT + b_kb + 8] = hi;
    }
    __syncthreads();

    // ---- fragments: contiguous LDS runs -> ds_load_b128 ----
    const int arow = (wm * 16 + r) * LDA;
    v8h a_lo = *(const v8h*)&As[arow + 8 * g];        // K = 8g..8g+7
    v8h a_hi = *(const v8h*)&As[arow + 16 + 8 * g];   // K = 16+8g..23+8g
    v16h af;
    #pragma unroll
    for (int i = 0; i < 8; ++i) { af[i] = a_lo[i]; af[8 + i] = a_hi[i]; }

    const int c0 = (wn * 32 + r) * LDBT;
    const int c1 = (wn * 32 + 16 + r) * LDBT;
    v8h b0l = *(const v8h*)&BsT[c0 + 8 * g];
    v8h b0h = *(const v8h*)&BsT[c0 + 16 + 8 * g];
    v8h b1l = *(const v8h*)&BsT[c1 + 8 * g];
    v8h b1h = *(const v8h*)&BsT[c1 + 16 + 8 * g];
    v16h bf0, bf1;
    #pragma unroll
    for (int i = 0; i < 8; ++i) {
      bf0[i] = b0l[i]; bf0[8 + i] = b0h[i];
      bf1[i] = b1l[i]; bf1[8 + i] = b1h[i];
    }

    acc0 = __builtin_amdgcn_wmma_f32_16x16x32_f16(false, af, false, bf0,
                                                  (short)0, acc0, false, false);
    acc1 = __builtin_amdgcn_wmma_f32_16x16x32_f16(false, af, false, bf1,
                                                  (short)0, acc1, false, false);
    __syncthreads();
  }

  // ---- epilogue: VGPR i -> row (i + 8g), col r of each 16x16 subtile ----
  const bool full = (rowBase + BM <= Mrows) && (colBase + BN <= Ncols);
  const int rr0 = rowBase + wm * 16 + 8 * g;
  const int cc0 = colBase + wn * 32 + r;
  if (full) {
    float bv0 = bias ? bias[cc0] : 0.0f;
    float bv1 = bias ? bias[cc0 + 16] : 0.0f;
    #pragma unroll
    for (int i = 0; i < 8; ++i) {
      float v0 = acc0[i] + bv0, v1 = acc1[i] + bv1;
      if (relu) { v0 = fmaxf(v0, 0.0f); v1 = fmaxf(v1, 0.0f); }
      size_t rowoff = (size_t)(rr0 + i) * Ncols;
      C[rowoff + cc0] = v0;
      C[rowoff + cc0 + 16] = v1;
    }
  } else {
    #pragma unroll
    for (int i = 0; i < 8; ++i) {
      int rr = rr0 + i;
      if (rr >= Mrows) continue;
      if (cc0 < Ncols) {
        float v = acc0[i] + (bias ? bias[cc0] : 0.0f);
        if (relu) v = fmaxf(v, 0.0f);
        C[(size_t)rr * Ncols + cc0] = v;
      }
      if (cc0 + 16 < Ncols) {
        float v = acc1[i] + (bias ? bias[cc0 + 16] : 0.0f);
        if (relu) v = fmaxf(v, 0.0f);
        C[(size_t)rr * Ncols + cc0 + 16] = v;
      }
    }
  }
}

// ---------------- host orchestration ----------------
static inline float* wsAlloc(char*& p, size_t nfloats) {
  float* r = (float*)p;
  p += ((nfloats * sizeof(float) + 255) / 256) * 256;
  return r;
}

extern "C" void kernel_launch(void* const* d_in, const int* in_sizes, int n_in,
                              void* d_out, int out_size, void* d_ws, size_t ws_size,
                              hipStream_t stream) {
  (void)n_in; (void)out_size; (void)ws_size;

  // ---- inputs (setup_inputs order) ----
  const float* X         = (const float*)d_in[0];
  const int*   H_rows    = (const int*)  d_in[1];
  const int*   H_cols    = (const int*)  d_in[2];
  const float* H_vals    = (const float*)d_in[3];
  const float* sub_X     = (const float*)d_in[4];
  const float* sub_e_lbl = (const float*)d_in[5];
  const int*   sH_rows   = (const int*)  d_in[6];
  const int*   sH_cols   = (const int*)  d_in[7];
  const float* sH_vals   = (const float*)d_in[8];
  const int*   sub_batch = (const int*)  d_in[9];
  const int*   sub_k_set = (const int*)  d_in[10];
  const int*   kh_rows   = (const int*)  d_in[11];
  const int*   kh_cols   = (const int*)  d_in[12];
  const float* kh_vals   = (const float*)d_in[13];
  const int*   g_batch   = (const int*)  d_in[14];

  // ---- params: detect dict flatten order via first W size ----
  int b_v2e, b_e2v, b_kv2e, b_ke2v, b_out;
  if (in_sizes[15] == FT * HID) {        // insertion order: mlps_v2e first
    b_v2e = 15; b_e2v = 27; b_kv2e = 39; b_ke2v = 51; b_out = 63;
  } else {                               // sorted keys: khwl_e2v first
    b_ke2v = 15; b_kv2e = 27; b_e2v = 39; b_v2e = 51; b_out = 63;
  }
  auto PW = [&](int base, int l, int j) { return (const float*)d_in[base + l * 4 + j * 2]; };
  auto PB = [&](int base, int l, int j) { return (const float*)d_in[base + l * 4 + j * 2 + 1]; };

  // ---- workspace layout ----
  char* wp = (char*)d_ws;
  float* Hv   = wsAlloc(wp, NNZ0);
  float* HTv  = wsAlloc(wp, NNZ0);
  float* sHv  = wsAlloc(wp, SUBNNZ);
  float* sHTv = wsAlloc(wp, SUBNNZ);
  float* kHv  = wsAlloc(wp, KHNNZ);
  float* kHTv = wsAlloc(wp, KHNNZ);
  float* deg  = wsAlloc(wp, SUBN);
  float* sub_he   = wsAlloc(wp, (size_t)SUBM * (FT + 1));
  float* sub_x    = wsAlloc(wp, (size_t)SUBN * (FT + 1));
  float* sub_pool = wsAlloc(wp, (size_t)SS * (FT + 1));
  float* nodeB[5]; for (int i = 0; i < 5; ++i) nodeB[i] = wsAlloc(wp, (size_t)NN * HID);
  float* edgeB[5]; for (int i = 0; i < 5; ++i) edgeB[i] = wsAlloc(wp, (size_t)MM * HID);
  float* sx   = wsAlloc(wp, (size_t)SS * SXP);    // padded concat buffer
  float* wpad = wsAlloc(wp, (size_t)SXP * HID);   // zero-padded khwl_v2e[0] W0
  float* o_h  = sub_he;                // S x 128 (sub_he dead by then)
  float* o    = sub_x;                 // S x 10
  float* sums = deg;                   // 64 x 10
  float* cnt  = deg + 1024;            // 64

  auto fill0 = [&](float* p, long n) {
    int blocks = (int)((n + 255) / 256); if (blocks > 8192) blocks = 8192;
    k_fill<<<blocks, 256, 0, stream>>>(p, 0.0f, n);
  };
  auto normPair = [&](const int* rows, const float* vals, float* out, int nrows, int nnz) {
    fill0(deg, nrows);
    k_degree<<<(nnz + 255) / 256, 256, 0, stream>>>(rows, vals, deg, nnz);
    k_normvals<<<(nnz + 255) / 256, 256, 0, stream>>>(rows, vals, deg, out, nnz);
  };
  auto gemm = [&](const float* A, int lda, const float* W, const float* b, float* C,
                  int Mr, int Nc, int Kd, int relu) {
    dim3 g((Nc + BN - 1) / BN, (Mr + BM - 1) / BM);
    k_gemm_wmma<<<g, 256, 0, stream>>>(A, lda, W, b, C, Mr, Nc, Kd, relu);
  };
  auto spmm = [&](const int* rows, const int* cols, const float* vals, const float* Xin,
                  int ldx, float* Out, int ldo, int F, int nnz) {
    k_spmm<<<nnz, 128, 0, stream>>>(rows, cols, vals, Xin, ldx, Out, ldo, F, nnz);
  };

  // ---- phase 0: row-normalize all six sparse operators ----
  normPair(H_rows,  H_vals,  Hv,   NN,   NNZ0);
  normPair(H_cols,  H_vals,  HTv,  MM,   NNZ0);
  normPair(sH_rows, sH_vals, sHv,  SUBN, SUBNNZ);
  normPair(sH_cols, sH_vals, sHTv, SUBM, SUBNNZ);
  normPair(kh_rows, kh_vals, kHv,  SS,   KHNNZ);
  normPair(kh_cols, kh_vals, kHTv, KHM,  KHNNZ);

  // ---- phase 1: sub-graph V2E -> +label -> E2V -> pool ----
  fill0(sub_he, (long)SUBM * (FT + 1));
  spmm(sH_cols, sH_rows, sHTv, sub_X, FT, sub_he, FT + 1, FT, SUBNNZ);
  k_setcol<<<(SUBM + 255) / 256, 256, 0, stream>>>(sub_he, FT + 1, FT, sub_e_lbl, SUBM);
  fill0(sub_x, (long)SUBN * (FT + 1));
  spmm(sH_rows, sH_cols, sHv, sub_he, FT + 1, sub_x, FT + 1, FT + 1, SUBNNZ);
  fill0(sub_pool, (long)SS * (FT + 1));
  k_segsum_rows<<<SUBN, 128, 0, stream>>>(sub_batch, sub_x, FT + 1, sub_pool, SUBN);

  // ---- phase 2: main hypergraph message passing ----
  float *xcur = nodeB[0], *xalt = nodeB[1];
  float *n_h = nodeB[2], *n_o = nodeB[3], *v_acc = nodeB[4];
  float *xecur = edgeB[0], *xealt = edgeB[1];
  float *e_h = edgeB[2], *e_o = edgeB[3], *e_acc = edgeB[4];
  const float* gin = X; int gk = FT;
  for (int idx = 0; idx < 3; ++idx) {
    gemm(gin, gk, PW(b_v2e, idx, 0), PB(b_v2e, idx, 0), n_h, NN, HID, gk, 1);
    gemm(n_h, HID, PW(b_v2e, idx, 1), PB(b_v2e, idx, 1), n_o, NN, HID, HID, 0);
    fill0(e_acc, (long)MM * HID);
    spmm(H_cols, H_rows, HTv, n_o, HID, e_acc, HID, HID, NNZ0);
    k_relu_add<<<8192, 256, 0, stream>>>(xealt, e_acc, idx ? xecur : nullptr, (long)MM * HID);
    { float* t = xecur; xecur = xealt; xealt = t; }
    gemm(xecur, HID, PW(b_e2v, idx, 0), PB(b_e2v, idx, 0), e_h, MM, HID, HID, 1);
    gemm(e_h,   HID, PW(b_e2v, idx, 1), PB(b_e2v, idx, 1), e_o, MM, HID, HID, 0);
    fill0(v_acc, (long)NN * HID);
    spmm(H_rows, H_cols, Hv, e_o, HID, v_acc, HID, HID, NNZ0);
    k_relu_add<<<8192, 256, 0, stream>>>(xalt, v_acc, idx ? xcur : nullptr, (long)NN * HID);
    { float* t = xcur; xcur = xalt; xalt = t; }
    gin = xcur; gk = HID;
  }

  // ---- phase 3: sx = concat(sub_pool[S,129], mean-gather x[S,256]), padded ----
  fill0(sx, (long)SS * SXP);   // zero pad columns 385..415
  k_copycols<<<SS, 128, 0, stream>>>(sx, SXP, 0, sub_pool, FT + 1, FT + 1, SS);
  k_gather_mean<<<SS, 256, 0, stream>>>(sub_k_set, xcur, HID, sx, SXP, FT + 1, HID, SS);
  // zero-padded copy of khwl_v2e[0] W0 (385x256 -> 416x256, extra K-rows = 0)
  fill0(wpad, (long)SXP * HID);
  k_copy<<<8192, 256, 0, stream>>>(wpad, PW(b_kv2e, 0, 0), (long)SXW * HID);

  // ---- phase 4: tuple-level hypergraph message passing (reuse buffers) ----
  float *kxcur = nodeB[0], *kxalt = nodeB[1];
  float *kn_h = nodeB[2], *kn_o = nodeB[3], *ks_acc = nodeB[4];
  float *kecur = edgeB[0], *kealt = edgeB[1];
  float *ke_h = edgeB[2], *ke_o = edgeB[3], *ke_acc = edgeB[4];
  for (int idx = 0; idx < 3; ++idx) {
    if (idx == 0) {
      // padded K: sx has lda=SXP, wpad has K=SXP rows; pad region is zero
      gemm(sx, SXP, wpad, PB(b_kv2e, 0, 0), kn_h, SS, HID, SXP, 1);
    } else {
      gemm(kxcur, HID, PW(b_kv2e, idx, 0), PB(b_kv2e, idx, 0), kn_h, SS, HID, HID, 1);
    }
    gemm(kn_h, HID, PW(b_kv2e, idx, 1), PB(b_kv2e, idx, 1), kn_o, SS, HID, HID, 0);
    fill0(ke_acc, (long)KHM * HID);
    spmm(kh_cols, kh_rows, kHTv, kn_o, HID, ke_acc, HID, HID, KHNNZ);
    k_relu_add<<<4096, 256, 0, stream>>>(kealt, ke_acc, idx ? kecur : nullptr, (long)KHM * HID);
    { float* t = kecur; kecur = kealt; kealt = t; }
    gemm(kecur, HID, PW(b_ke2v, idx, 0), PB(b_ke2v, idx, 0), ke_h, KHM, HID, HID, 1);
    gemm(ke_h,  HID, PW(b_ke2v, idx, 1), PB(b_ke2v, idx, 1), ke_o, KHM, HID, HID, 0);
    fill0(ks_acc, (long)SS * HID);
    spmm(kh_rows, kh_cols, kHv, ke_o, HID, ks_acc, HID, HID, KHNNZ);
    k_relu_add<<<4096, 256, 0, stream>>>(kxalt, ks_acc, idx ? kxcur : nullptr, (long)SS * HID);
    { float* t = kxcur; kxcur = kxalt; kxalt = t; }
  }

  // ---- phase 5: out MLP (256 -> 128 -> 10) ----
  gemm(kxcur, HID, PW(b_out, 0, 0), PB(b_out, 0, 0), o_h, SS, HID / 2, HID, 1);
  gemm(o_h, HID / 2, PW(b_out, 0, 1), PB(b_out, 0, 1), o, SS, NCLS, HID / 2, 0);

  // ---- phase 6: per-graph mean readout ----
  fill0(sums, NG * NCLS);
  fill0(cnt, NG);
  k_segsum_rows<<<SS, 32, 0, stream>>>(g_batch, o, NCLS, sums, SS);
  k_count<<<(SS + 255) / 256, 256, 0, stream>>>(g_batch, cnt, SS);
  k_div<<<(NG * NCLS + 255) / 256, 256, 0, stream>>>((float*)d_out, sums, cnt, NG, NCLS);
}
